// SelfAttention_20323785245050
// MI455X (gfx1250) — matrix operand order repaired and checked by reading, not compiled, not run on hardware
//
#include <hip/hip_runtime.h>

// ---------------------------------------------------------------------------
// Causal self-attention for MI455X (gfx1250, wave32, WMMA).
//   B=4, S=2048, D_MODEL=1024, HEAD_DIM=64, fp32 in/out.
// Compute-bound (AI ~215 FLOP/byte) -> v_wmma_f32_16x16x32_bf16 (bf16 in,
// fp32 accumulate), fp32 online softmax.  Kernel 2 double-buffers K/V^T
// tiles in LDS with GLOBAL_LOAD_ASYNC_TO_LDS_B128 (ASYNCcnt) to overlap
// memory with the XDL pipe.
// ---------------------------------------------------------------------------

#define B_ 4
#define S_ 2048
#define D_ 1024
#define H_ 64

typedef __attribute__((ext_vector_type(16))) __bf16        v16bf;
typedef __attribute__((ext_vector_type(8)))  float         v8f;
typedef __attribute__((ext_vector_type(4)))  unsigned int  u32x4;

union FragB16 {
    v16bf v;
    u32x4 q[2];
};

// Async copy 16B global -> LDS; INST_OFFSET is added to BOTH addresses
// (CDNA5 ISA), so one base covers strided tiles.
#define ASYNC_LD_B128(ldsaddr, gaddr, off)                                  \
    asm volatile("global_load_async_to_lds_b128 %0, %1, off offset:" #off   \
                 :: "v"(ldsaddr), "v"(gaddr) : "memory")
#define WAIT_ASYNC(n) asm volatile("s_wait_asynccnt " #n ::: "memory")

// Low 32 bits of a generic pointer to LDS == LDS byte offset (flat-LDS
// aperture mapping discards the upper bits, CDNA5 ISA).
__device__ __forceinline__ unsigned lds_addr32(const void* p) {
    return (unsigned)(unsigned long long)p;
}

// fp32 -> bf16 (round-to-nearest-even); inputs here are never NaN.
__device__ __forceinline__ unsigned short f2bf(float f) {
    unsigned int u = __float_as_uint(f);
    u += 0x7FFFu + ((u >> 16) & 1u);
    return (unsigned short)(u >> 16);
}
__device__ __forceinline__ unsigned int pack2(float lo, float hi) {
    return (unsigned int)f2bf(lo) | ((unsigned int)f2bf(hi) << 16);
}

// ---------------------------------------------------------------------------
// Kernel 1: Q/K/V projections.  [B*S,1024] x [1024,64] -> 3x [B*S,64]
// 64 rows per block, 128 threads = 4 waves; each wave does a 16x64 tile of
// Q, K and V (12 wmma accumulators).  Q pre-scaled by 1/sqrt(64)=0.125.
// V written transposed ([b][h][s]) for the PV matmul's B-fragments.
// ---------------------------------------------------------------------------
__global__ __launch_bounds__(128) void qkv_proj_kernel(
    const float* __restrict__ x,  const float* __restrict__ Wq,
    const float* __restrict__ Wk, const float* __restrict__ Wv,
    unsigned short* __restrict__ Qb, unsigned short* __restrict__ Kb,
    unsigned short* __restrict__ Vt)
{
    __shared__ __align__(16) unsigned short xs[64][32];     // x tile, bf16, [row][k]
    __shared__ __align__(16) unsigned short ws[3][64][32];  // W^T tiles, bf16, [n][k]

    const int tid     = threadIdx.x;
    const int lane    = tid & 31;
    const int w       = tid >> 5;
    const int nl      = lane & 15;
    const int rowbase = blockIdx.x * 64;            // flat row in [0, B*S)

    const float* Wm[3] = { Wq, Wk, Wv };

    v8f acc[3][4];
    for (int m = 0; m < 3; ++m)
        for (int nt = 0; nt < 4; ++nt)
            for (int e = 0; e < 8; ++e) acc[m][nt][e] = 0.0f;

    for (int kt = 0; kt < D_ / 32; ++kt) {
        const int k0 = kt * 32;

        // ---- stage x tile (fp32 -> bf16), coalesced ----
        {
            const int row = tid >> 1;
            const int kk  = (tid & 1) * 16;
            const float* src = x + (size_t)(rowbase + row) * D_ + k0 + kk;
            if (kt + 1 < D_ / 32) __builtin_prefetch(src + 32, 0, 1);
            float f[16];
            #pragma unroll
            for (int i = 0; i < 16; ++i) f[i] = src[i];
            u32x4 a, b;
            a[0] = pack2(f[0],  f[1]);  a[1] = pack2(f[2],  f[3]);
            a[2] = pack2(f[4],  f[5]);  a[3] = pack2(f[6],  f[7]);
            b[0] = pack2(f[8],  f[9]);  b[1] = pack2(f[10], f[11]);
            b[2] = pack2(f[12], f[13]); b[3] = pack2(f[14], f[15]);
            *(u32x4*)&xs[row][kk]     = a;
            *(u32x4*)&xs[row][kk + 8] = b;
        }
        // ---- stage W tiles transposed (read along n coalesced, write [n][k]) ----
        {
            const int kk = tid >> 2;
            const int n0 = (tid & 3) * 16;
            #pragma unroll
            for (int m = 0; m < 3; ++m) {
                const float* src = Wm[m] + (size_t)(k0 + kk) * H_ + n0;
                if (kt + 1 < D_ / 32) __builtin_prefetch(src + 32 * H_, 0, 1);
                #pragma unroll
                for (int j = 0; j < 16; ++j) ws[m][n0 + j][kk] = f2bf(src[j]);
            }
        }
        __syncthreads();

        // ---- wave-level 16x64x32 WMMA step ----
        // Load ALL fragments first, then issue 12 independent WMMAs
        // back-to-back (keeps the XDL pipe fed, one dscnt wait).
        {
            const int arow = w * 16 + nl;
            const int kbA  = (lane >> 4) * 8;     // A frag: k [kbA, kbA+8) and [kbA+16, kbA+24)
            const int kbB  = (lane >> 4) * 8;     // B frag: k [kbB, kbB+8) and [kbB+16, kbB+24)
            FragB16 a;
            a.q[0] = *(const u32x4*)&xs[arow][kbA];
            a.q[1] = *(const u32x4*)&xs[arow][kbA + 16];
            FragB16 fb[12];
            #pragma unroll
            for (int m = 0; m < 3; ++m) {
                #pragma unroll
                for (int nt = 0; nt < 4; ++nt) {
                    const int bn = nt * 16 + nl;
                    fb[m * 4 + nt].q[0] = *(const u32x4*)&ws[m][bn][kbB];
                    fb[m * 4 + nt].q[1] = *(const u32x4*)&ws[m][bn][kbB + 16];
                }
            }
            #pragma unroll
            for (int m = 0; m < 3; ++m) {
                #pragma unroll
                for (int nt = 0; nt < 4; ++nt) {
                    acc[m][nt] = __builtin_amdgcn_wmma_f32_16x16x32_bf16(
                        false, a.v, false, fb[m * 4 + nt].v, (short)0,
                        acc[m][nt], false, false);
                }
            }
        }
        __syncthreads();
    }

    // ---- epilogue: C layout lane l, vgpr r -> row (l<16? r : r+8), col l%16 ----
    #pragma unroll
    for (int nt = 0; nt < 4; ++nt) {
        const int n = nt * 16 + nl;
        #pragma unroll
        for (int r = 0; r < 8; ++r) {
            const int M  = (lane < 16) ? r : r + 8;
            const int gr = rowbase + w * 16 + M;            // flat row b*S + s
            Qb[(size_t)gr * H_ + n] = f2bf(acc[0][nt][r] * 0.125f);
            Kb[(size_t)gr * H_ + n] = f2bf(acc[1][nt][r]);
            const int bb = gr / S_;
            const int s  = gr - bb * S_;
            Vt[(size_t)bb * H_ * S_ + (size_t)n * S_ + s] = f2bf(acc[2][nt][r]);
        }
    }
}

// ---------------------------------------------------------------------------
// Kernel 2: causal flash attention.  One block per (batch, 64-query tile),
// 128 threads = 4 waves, each wave owns 16 query rows.  fp32 online softmax.
// K/V^T tiles double-buffered in LDS via async global->LDS copies.
// ---------------------------------------------------------------------------
__global__ __launch_bounds__(128) void flash_attn_kernel(
    const unsigned short* __restrict__ Qb, const unsigned short* __restrict__ Kb,
    const unsigned short* __restrict__ Vt, float* __restrict__ out)
{
    __shared__ __align__(16) unsigned short kt_s[2][64][64];   // K tile   [kpos][h]
    __shared__ __align__(16) unsigned short vt_s[2][64][64];   // V^T tile [h][kpos]
    __shared__ __align__(16) unsigned short p_s[4][16][64];    // P tile per wave

    const int tid   = threadIdx.x;
    const int lane  = tid & 31;
    const int w     = tid >> 5;
    const int nl    = lane & 15;
    const int kbA   = (lane >> 4) * 8;
    const int kbB   = (lane >> 4) * 8;
    const int qt    = blockIdx.x;
    const int b     = blockIdx.y;
    const int qbase = qt * 64;
    const float NEG_INF = -__builtin_inff();

    // Per-thread staging assignment (same every tile): K tile chunk and V^T row.
    const int vh    = tid >> 1;               // V^T row (head dim)
    const int vhalf = (tid & 1) * 32;         // 32-key half of that row
    const unsigned ldsK0 = lds_addr32(&kt_s[0][0][0]) + (unsigned)(tid * 16);
    const unsigned ldsK1 = lds_addr32(&kt_s[1][0][0]) + (unsigned)(tid * 16);
    const unsigned ldsV0 = lds_addr32(&vt_s[0][vh][vhalf]);
    const unsigned ldsV1 = lds_addr32(&vt_s[1][vh][vhalf]);
    const unsigned short* gK0 = Kb + (size_t)(b * S_) * H_ + tid * 8;
    const unsigned short* gV0 = Vt + (size_t)b * H_ * S_ + (size_t)vh * S_ + vhalf;

    // Issue one tile's async loads: 4x16B for K (stride 2KB both sides),
    // 4x16B for V^T row chunk (contiguous).  8 ASYNCcnt per wave per tile.
    #define ISSUE_TILE(jj, ldsK, ldsV)                                         \
        do {                                                                   \
            const unsigned short* gk = gK0 + (size_t)(jj) * 64 * H_;           \
            const unsigned short* gv = gV0 + (size_t)(jj) * 64;                \
            ASYNC_LD_B128(ldsK, gk, 0);                                        \
            ASYNC_LD_B128(ldsK, gk, 2048);                                     \
            ASYNC_LD_B128(ldsK, gk, 4096);                                     \
            ASYNC_LD_B128(ldsK, gk, 6144);                                     \
            ASYNC_LD_B128(ldsV, gv, 0);                                        \
            ASYNC_LD_B128(ldsV, gv, 16);                                       \
            ASYNC_LD_B128(ldsV, gv, 32);                                       \
            ASYNC_LD_B128(ldsV, gv, 48);                                       \
        } while (0)

    // Q A-fragments (already scaled by 1/sqrt(H)); head_dim 64 = 2 k-steps
    FragB16 qf[2];
    {
        const int qrow = qbase + w * 16 + nl;
        const unsigned short* qp = Qb + (size_t)(b * S_ + qrow) * H_;
        #pragma unroll
        for (int ks = 0; ks < 2; ++ks) {
            qf[ks].q[0] = *(const u32x4*)(qp + ks * 32 + kbA);
            qf[ks].q[1] = *(const u32x4*)(qp + ks * 32 + kbA + 16);
        }
    }

    v8f   o[4];
    float m8[8], l8[8];
    for (int nt = 0; nt < 4; ++nt)
        for (int e = 0; e < 8; ++e) o[nt][e] = 0.0f;
    for (int r = 0; r < 8; ++r) { m8[r] = NEG_INF; l8[r] = 0.0f; }

    ISSUE_TILE(0, ldsK0, ldsV0);                 // prologue: tile 0 -> buffer 0

    for (int j = 0; j <= qt; ++j) {
        const int buf = j & 1;
        // Prefetch tile j+1 into the other buffer (read-free since the
        // end-of-iteration barrier of iter j-1), then wait for tile j:
        // async loads complete in order, so <=8 outstanding => tile j landed.
        if (j < qt) {
            if (buf == 0) ISSUE_TILE(j + 1, ldsK1, ldsV1);
            else          ISSUE_TILE(j + 1, ldsK0, ldsV0);
            WAIT_ASYNC(8);
        } else {
            WAIT_ASYNC(0);
        }
        __syncthreads();                         // tile j visible to all waves

        const int kpos0 = j * 64;

        // ---- S = Q K^T : load all 8 B-fragments, then 8 WMMAs ----
        v8f sacc[4];
        for (int nt = 0; nt < 4; ++nt)
            for (int e = 0; e < 8; ++e) sacc[nt][e] = 0.0f;
        {
            FragB16 bk[8];
            #pragma unroll
            for (int ks = 0; ks < 2; ++ks) {
                #pragma unroll
                for (int nt = 0; nt < 4; ++nt) {
                    const int kp = nt * 16 + nl;         // key position = column
                    bk[ks * 4 + nt].q[0] = *(const u32x4*)&kt_s[buf][kp][ks * 32 + kbB];
                    bk[ks * 4 + nt].q[1] = *(const u32x4*)&kt_s[buf][kp][ks * 32 + kbB + 16];
                }
            }
            #pragma unroll
            for (int ks = 0; ks < 2; ++ks)
                #pragma unroll
                for (int nt = 0; nt < 4; ++nt)
                    sacc[nt] = __builtin_amdgcn_wmma_f32_16x16x32_bf16(
                        false, qf[ks].v, false, bk[ks * 4 + nt].v, (short)0,
                        sacc[nt], false, false);
        }

        // ---- causal mask (only the diagonal tile is partial) ----
        if (j == qt) {
            #pragma unroll
            for (int nt = 0; nt < 4; ++nt) {
                const int kg = kpos0 + nt * 16 + nl;
                #pragma unroll
                for (int r = 0; r < 8; ++r) {
                    const int qg = qbase + w * 16 + ((lane < 16) ? r : r + 8);
                    if (kg > qg) sacc[nt][r] = NEG_INF;
                }
            }
        }

        // ---- online softmax (per-row stats; rows live in 16-lane halves) ----
        #pragma unroll
        for (int r = 0; r < 8; ++r) {
            float mx = fmaxf(fmaxf(sacc[0][r], sacc[1][r]),
                             fmaxf(sacc[2][r], sacc[3][r]));
            #pragma unroll
            for (int msk = 1; msk < 16; msk <<= 1)
                mx = fmaxf(mx, __shfl_xor(mx, msk, 32));
            const float mn    = fmaxf(m8[r], mx);
            const float alpha = __expf(m8[r] - mn);      // 0 on first tile
            float rs = 0.0f;
            #pragma unroll
            for (int nt = 0; nt < 4; ++nt) {
                const float p = __expf(sacc[nt][r] - mn);
                sacc[nt][r] = p;
                rs += p;
            }
            #pragma unroll
            for (int msk = 1; msk < 16; msk <<= 1)
                rs += __shfl_xor(rs, msk, 32);
            l8[r] = l8[r] * alpha + rs;
            m8[r] = mn;
            #pragma unroll
            for (int nt = 0; nt < 4; ++nt) o[nt][r] *= alpha;
        }

        // ---- restage P (C layout) -> LDS -> A-fragment layout ----
        #pragma unroll
        for (int nt = 0; nt < 4; ++nt) {
            #pragma unroll
            for (int r = 0; r < 8; ++r) {
                const int M = (lane < 16) ? r : r + 8;
                p_s[w][M][nt * 16 + nl] = f2bf(sacc[nt][r]);
            }
        }
        __syncthreads();   // uniform; orders b16 stores vs b128 reloads

        // ---- O += P V : load all fragments, then 8 WMMAs ----
        {
            FragB16 pa[2], bv[8];
            #pragma unroll
            for (int ks = 0; ks < 2; ++ks) {
                pa[ks].q[0] = *(const u32x4*)&p_s[w][nl][ks * 32 + kbA];
                pa[ks].q[1] = *(const u32x4*)&p_s[w][nl][ks * 32 + kbA + 16];
                #pragma unroll
                for (int nt = 0; nt < 4; ++nt) {
                    const int h = nt * 16 + nl;          // output head col
                    bv[ks * 4 + nt].q[0] = *(const u32x4*)&vt_s[buf][h][ks * 32 + kbB];
                    bv[ks * 4 + nt].q[1] = *(const u32x4*)&vt_s[buf][h][ks * 32 + kbB + 16];
                }
            }
            #pragma unroll
            for (int ks = 0; ks < 2; ++ks)
                #pragma unroll
                for (int nt = 0; nt < 4; ++nt)
                    o[nt] = __builtin_amdgcn_wmma_f32_16x16x32_bf16(
                        false, pa[ks].v, false, bv[ks * 4 + nt].v, (short)0,
                        o[nt], false, false);
        }
        __syncthreads();   // all waves done reading buf before it is re-filled
    }

    // ---- normalize by l and store fp32 output ----
    #pragma unroll
    for (int nt = 0; nt < 4; ++nt) {
        const int h = nt * 16 + nl;
        #pragma unroll
        for (int r = 0; r < 8; ++r) {
            const int M  = (lane < 16) ? r : r + 8;
            const int qg = qbase + w * 16 + M;
            out[(size_t)(b * S_ + qg) * H_ + h] = o[nt][r] / l8[r];
        }
    }
    #undef ISSUE_TILE
}

// ---------------------------------------------------------------------------
// Workspace layout (bf16): Qb [B*S][64] | Kb [B*S][64] | Vt [B][64][S]
// = 3 MB total.
// ---------------------------------------------------------------------------
extern "C" void kernel_launch(void* const* d_in, const int* in_sizes, int n_in,
                              void* d_out, int out_size, void* d_ws, size_t ws_size,
                              hipStream_t stream) {
    const float* x  = (const float*)d_in[0];
    const float* Wq = (const float*)d_in[1];
    const float* Wk = (const float*)d_in[2];
    const float* Wv = (const float*)d_in[3];
    // d_in[4] is the causal mask; applied analytically (k <= q) instead.

    unsigned short* Qb = (unsigned short*)d_ws;
    unsigned short* Kb = Qb + (size_t)B_ * S_ * H_;
    unsigned short* Vt = Kb + (size_t)B_ * S_ * H_;

    qkv_proj_kernel<<<dim3((B_ * S_) / 64), dim3(128), 0, stream>>>(
        x, Wq, Wk, Wv, Qb, Kb, Vt);
    flash_attn_kernel<<<dim3(S_ / 64, B_), dim3(128), 0, stream>>>(
        Qb, Kb, Vt, (float*)d_out);
}
